// XLSTM_11871289606794
// MI455X (gfx1250) — matrix operand order repaired and checked by reading, not compiled, not run on hardware
//
#include <hip/hip_runtime.h>
#include <stdint.h>

#define S_LEN 512
#define BATCH 64
#define DIN   512
#define HDIM  512
#define EPSLN 1e-5f

typedef __attribute__((ext_vector_type(16))) __bf16 v16bf;
typedef __attribute__((ext_vector_type(8)))  float  v8f;

union Frag16 {
    v16bf v;
    uint4 q[2];
    unsigned short u[16];
};

__device__ __forceinline__ unsigned short f2bf(float f) {
    unsigned int u = __float_as_uint(f);
    unsigned int r = u + 0x7FFFu + ((u >> 16) & 1u);   // RNE
    return (unsigned short)(r >> 16);
}
__device__ __forceinline__ float bf2f(unsigned short h) {
    return __uint_as_float(((unsigned int)h) << 16);
}

// ---------------------------------------------------------------------------
// Packed weight fragment layout (B-matrix, 32x16 bf16 per fragment):
//   element e of lane l  ->  W[kb*32 + (l>>4)*8 + e + (e>=8 ? 8 : 0)][nb*16 + (l&15)]
//   fragment = 1024 bytes = 32 lanes * 32 contiguous bytes per lane
// Fragment index space: mat*512 + kb*32 + nb   (mat: 0-4 Wx, 5-9 Wh, 10-13 Wm, 14 Wpm)
// ---------------------------------------------------------------------------
__global__ __launch_bounds__(256) void pack_weights_kernel(
    const float* __restrict__ Wx, const float* __restrict__ Wh,
    const float* __restrict__ Wm, const float* __restrict__ Wpm,
    unsigned short* __restrict__ pack)
{
    int gid  = blockIdx.x * 256 + threadIdx.x;   // 15*16*32*32 = 245760 threads
    int lane = gid & 31; int idx = gid >> 5;
    int nb   = idx & 31; idx >>= 5;
    int kb   = idx & 15; idx >>= 4;
    int mat  = idx;                               // 0..14
    const float* W;
    if      (mat < 5)  W = Wx  + (size_t)mat        * (DIN  * HDIM);
    else if (mat < 10) W = Wh  + (size_t)(mat - 5)  * (HDIM * HDIM);
    else if (mat < 14) W = Wm  + (size_t)(mat - 10) * (HDIM * HDIM);
    else               W = Wpm;

    int kbase = kb * 32 + (lane >> 4) * 8;
    int n     = nb * 16 + (lane & 15);
    union { unsigned short u[16]; uint4 q[2]; } t;
#pragma unroll
    for (int e = 0; e < 16; ++e)
        t.u[e] = f2bf(W[(size_t)(kbase + e + (e >= 8 ? 8 : 0)) * HDIM + n]);

    size_t frag = (size_t)mat * 512 + kb * 32 + nb;
    uint4* dst = (uint4*)(pack + frag * 512 + lane * 16);  // ushort units
    dst[0] = t.q[0];
    dst[1] = t.q[1];
}

// ---------------------------------------------------------------------------
// xp = x @ Wx, stored bf16 in C-tile layout (8 bf16 per lane per tile).
// Tile index: (t*4+btile)*160 + g*32 + nb ; tile = 512 bytes.
// ---------------------------------------------------------------------------
__global__ __launch_bounds__(256) void xproj_kernel(
    const float* __restrict__ x, const unsigned short* __restrict__ pack,
    unsigned short* __restrict__ xp)
{
    int t     = blockIdx.x >> 2;
    int btile = blockIdx.x & 3;
    int lane  = threadIdx.x & 31;
    int wave  = threadIdx.x >> 5;
    int khi   = (lane >> 4) * 8;

    // Load all 16 A-fragments of this (t,btile) x-tile (16 rows x 512 K) as bf16.
    Frag16 af[16];
    const float* xrow = x + ((size_t)t * BATCH + btile * 16 + (lane & 15)) * DIN;
#pragma unroll
    for (int kb = 0; kb < 16; ++kb) {
        const float* p0 = xrow + kb * 32 + khi;
#pragma unroll
        for (int e = 0; e < 8; ++e) {
            af[kb].u[e]     = f2bf(p0[e]);
            af[kb].u[8 + e] = f2bf(p0[16 + e]);
        }
    }

    for (int tile = wave; tile < 160; tile += 8) {
        int g = tile >> 5, nb = tile & 31;
        v8f acc0, acc1;
#pragma unroll
        for (int i = 0; i < 8; ++i) { acc0[i] = 0.f; acc1[i] = 0.f; }
        // Two independent accumulation chains to keep the matrix pipe busy.
#pragma unroll
        for (int kb = 0; kb < 16; kb += 2) {
            size_t frag0 = (size_t)g * 512 + kb * 32 + nb;
            const uint4* bp0 = (const uint4*)(pack + frag0 * 512 + lane * 16);
            const uint4* bp1 = (const uint4*)(pack + (frag0 + 32) * 512 + lane * 16);
            if (kb < 14)
                __builtin_prefetch((const void*)(bp0 + 128), 0, 1);
            Frag16 b0, b1;
            b0.q[0] = bp0[0]; b0.q[1] = bp0[1];
            b1.q[0] = bp1[0]; b1.q[1] = bp1[1];
            acc0 = __builtin_amdgcn_wmma_f32_16x16x32_bf16(
                false, af[kb].v, false, b0.v, (short)0, acc0, false, false);
            acc1 = __builtin_amdgcn_wmma_f32_16x16x32_bf16(
                false, af[kb + 1].v, false, b1.v, (short)0, acc1, false, false);
        }
        union { unsigned short u[8]; uint4 q; } o;
#pragma unroll
        for (int i = 0; i < 8; ++i) o.u[i] = f2bf(acc0[i] + acc1[i]);
        size_t tileIdx = (size_t)(t * 4 + btile) * 160 + tile;
        *(uint4*)(xp + tileIdx * 256 + lane * 8) = o.q;
    }
}

// ---------------------------------------------------------------------------
// Persistent recurrence: 4 workgroups (one per 16-row batch tile), 16 waves.
// Wave w owns column blocks {2w, 2w+1} of H. State tiles (bf16 16x512) in LDS.
// Loop order: one A-fragment load feeds 5 (or 4) independent WMMA chains.
// ---------------------------------------------------------------------------
__global__ __launch_bounds__(512) void recurrent_kernel(
    const unsigned short* __restrict__ xp,
    const unsigned short* __restrict__ pack,
    const float* __restrict__ bx,  const float* __restrict__ bh,
    const float* __restrict__ bm,  const float* __restrict__ bpm,
    const float* __restrict__ g_c, const float* __restrict__ b_c,
    const float* __restrict__ g_m, const float* __restrict__ b_m,
    const float* __restrict__ g_h, const float* __restrict__ b_h,
    float* __restrict__ out)
{
    __shared__ alignas(16) unsigned short h_bf[16 * 512];
    __shared__ alignas(16) unsigned short m_bf[16 * 512];
    __shared__ alignas(16) unsigned short c_bf[16 * 512];
    __shared__ float lnS[6][16];   // cSum,cSq, mSum,mSq, hSum,hSq per row

    const int btile = blockIdx.x;
    const int tid   = threadIdx.x;
    const int lane  = tid & 31;
    const int wave  = tid >> 5;
    const int mlow  = lane & 15;
    const int g16   = lane >> 4;       // 0 / 1
    const int khi   = g16 * 8;

    for (int i = tid; i < 16 * 512; i += 512) { h_bf[i] = 0; m_bf[i] = 0; c_bf[i] = 0; }
    if (tid < 96) lnS[tid >> 4][tid & 15] = 0.f;
    __syncthreads();

    // Per-owned-column constants (t-invariant).
    const int nbs[2] = { wave * 2, wave * 2 + 1 };
    float biasg[2][5], bpmv[2], gcv[2], bcv[2], gmv[2], bmv[2], ghv[2], bhv[2];
#pragma unroll
    for (int j = 0; j < 2; ++j) {
        int n = nbs[j] * 16 + mlow;
#pragma unroll
        for (int g = 0; g < 4; ++g)
            biasg[j][g] = bx[g * HDIM + n] + bh[g * HDIM + n] + bm[g * HDIM + n];
        biasg[j][4] = bx[4 * HDIM + n] + bh[4 * HDIM + n];
        bpmv[j] = bpm[n];
        gcv[j] = g_c[n]; bcv[j] = b_c[n];
        gmv[j] = g_m[n]; bmv[j] = b_m[n];
        ghv[j] = g_h[n]; bhv[j] = b_h[n];
    }

    auto loadA = [&](const unsigned short* buf, int kb) -> v16bf {
        Frag16 a;
        const unsigned short* p = buf + mlow * 512 + kb * 32 + khi;
        a.q[0] = *(const uint4*)p;          // K run 0..7
        a.q[1] = *(const uint4*)(p + 16);   // K run 16..23
        return a.v;
    };
    auto loadB = [&](int frag) -> v16bf {
        Frag16 b;
        const uint4* p = (const uint4*)(pack + (size_t)frag * 512 + lane * 16);
        b.q[0] = p[0]; b.q[1] = p[1];
        return b.v;
    };
    auto lnAccum = [&](float* sum, float* sq, const float* v8) {
#pragma unroll
        for (int r = 0; r < 8; ++r) {
            float s = v8[r], q = v8[r] * v8[r];
#pragma unroll
            for (int m = 1; m < 16; m <<= 1) {
                s += __shfl_xor(s, m, 16);
                q += __shfl_xor(q, m, 16);
            }
            if (mlow == 0) {
                atomicAdd(&sum[r + 8 * g16], s);
                atomicAdd(&sq [r + 8 * g16], q);
            }
        }
    };

    const int FR_WH = 2560, FR_WM = 5120, FR_WPM = 7168;
    float o_g[2][8], e_g[2][8], cn[2][8], mr[2][8], hp[2][8];

    for (int t = 0; t < S_LEN; ++t) {
        const size_t xpTileBase = (size_t)(t * 4 + btile) * 160;

        // ---- Phase 1: gate pre-activations + c_new, partial LN sums ----
#pragma unroll
        for (int j = 0; j < 2; ++j) {
            const int nb = nbs[j];
            // 5 independent accumulator chains, init from xp tiles.
            v8f accs[5];
#pragma unroll
            for (int g = 0; g < 5; ++g) {
                Frag16 xf;
                xf.q[0] = *(const uint4*)(xp + (xpTileBase + g * 32 + nb) * 256 + lane * 8);
#pragma unroll
                for (int i = 0; i < 8; ++i) accs[g][i] = bf2f(xf.u[i]);
            }
            // h @ Wh : one A-fragment feeds 5 chains.
#pragma unroll
            for (int kb = 0; kb < 16; ++kb) {
                v16bf a = loadA(h_bf, kb);
                if (kb < 15)
                    __builtin_prefetch(
                        (const void*)(pack + (size_t)(FR_WH + (kb + 1) * 32 + nb) * 512 + lane * 16),
                        0, 1);
#pragma unroll
                for (int g = 0; g < 5; ++g) {
                    v16bf b = loadB(FR_WH + g * 512 + kb * 32 + nb);
                    accs[g] = __builtin_amdgcn_wmma_f32_16x16x32_bf16(
                        false, a, false, b, (short)0, accs[g], false, false);
                }
            }
            // m @ Wm : one A-fragment feeds 4 chains.
#pragma unroll
            for (int kb = 0; kb < 16; ++kb) {
                v16bf a = loadA(m_bf, kb);
#pragma unroll
                for (int g = 0; g < 4; ++g) {
                    v16bf b = loadB(FR_WM + g * 512 + kb * 32 + nb);
                    accs[g] = __builtin_amdgcn_wmma_f32_16x16x32_bf16(
                        false, a, false, b, (short)0, accs[g], false, false);
                }
            }
            // Activations.
            float ig[8], fg[8], ctg[8];
#pragma unroll
            for (int i = 0; i < 8; ++i) {
                ig[i]     = 1.f / (1.f + __expf(-(accs[0][i] + biasg[j][0])));
                fg[i]     = 1.f / (1.f + __expf(-(accs[1][i] + biasg[j][1])));
                o_g[j][i] = 1.f / (1.f + __expf(-(accs[2][i] + biasg[j][2])));
                e_g[j][i] = 1.f / (1.f + __expf(-(accs[3][i] + biasg[j][3])));
                ctg[i]    = tanhf(accs[4][i] + biasg[j][4]);
            }
#pragma unroll
            for (int i = 0; i < 8; ++i) {
                int row = i + 8 * g16;
                float cold = bf2f(c_bf[row * 512 + nb * 16 + mlow]);
                cn[j][i] = fg[i] * cold + ig[i] * ctg[i];
            }
            lnAccum(lnS[0], lnS[1], cn[j]);
        }
        __syncthreads();

        // ---- Phase 2: LN(c) -> cbar, publish bf16 state ----
#pragma unroll
        for (int j = 0; j < 2; ++j) {
            const int nb = nbs[j];
#pragma unroll
            for (int i = 0; i < 8; ++i) {
                int row = i + 8 * g16;
                float mu  = lnS[0][row] * (1.f / HDIM);
                float var = lnS[1][row] * (1.f / HDIM) - mu * mu;
                float rs  = rsqrtf(var + EPSLN);
                float cb  = (cn[j][i] - mu) * rs * gcv[j] + bcv[j];
                cn[j][i]  = cb;                                  // keep cbar
                c_bf[row * 512 + nb * 16 + mlow] = f2bf(cb);
            }
        }
        __syncthreads();

        // ---- Phase 3: pm = cbar @ Wpm ; m_raw ; partial LN sums ----
#pragma unroll
        for (int j = 0; j < 2; ++j) {
            const int nb = nbs[j];
            v8f acc0, acc1;
#pragma unroll
            for (int i = 0; i < 8; ++i) { acc0[i] = 0.f; acc1[i] = 0.f; }
#pragma unroll
            for (int kb = 0; kb < 16; kb += 2) {
                v16bf a0 = loadA(c_bf, kb);
                v16bf a1 = loadA(c_bf, kb + 1);
                v16bf b0 = loadB(FR_WPM + kb * 32 + nb);
                v16bf b1 = loadB(FR_WPM + (kb + 1) * 32 + nb);
                acc0 = __builtin_amdgcn_wmma_f32_16x16x32_bf16(
                    false, a0, false, b0, (short)0, acc0, false, false);
                acc1 = __builtin_amdgcn_wmma_f32_16x16x32_bf16(
                    false, a1, false, b1, (short)0, acc1, false, false);
            }
#pragma unroll
            for (int i = 0; i < 8; ++i) {
                int row = i + 8 * g16;
                float mold = bf2f(m_bf[row * 512 + nb * 16 + mlow]);
                float e    = e_g[j][i];
                mr[j][i] = (1.f - e) * mold + e * (acc0[i] + acc1[i] + bpmv[j]);
            }
            lnAccum(lnS[2], lnS[3], mr[j]);
        }
        __syncthreads();

        // ---- Phase 4: LN(m) -> mbar ; h_pre = o*tanh(cbar+mbar) ----
#pragma unroll
        for (int j = 0; j < 2; ++j) {
            const int nb = nbs[j];
#pragma unroll
            for (int i = 0; i < 8; ++i) {
                int row = i + 8 * g16;
                float mu  = lnS[2][row] * (1.f / HDIM);
                float var = lnS[3][row] * (1.f / HDIM) - mu * mu;
                float rs  = rsqrtf(var + EPSLN);
                float mb  = (mr[j][i] - mu) * rs * gmv[j] + bmv[j];
                m_bf[row * 512 + nb * 16 + mlow] = f2bf(mb);
                hp[j][i] = o_g[j][i] * tanhf(cn[j][i] + mb);
            }
            lnAccum(lnS[4], lnS[5], hp[j]);
        }
        __syncthreads();

        // ---- Phase 5: LN(h) -> hbar ; publish h state + output ----
#pragma unroll
        for (int j = 0; j < 2; ++j) {
            const int nb = nbs[j];
#pragma unroll
            for (int i = 0; i < 8; ++i) {
                int row = i + 8 * g16;
                float mu  = lnS[4][row] * (1.f / HDIM);
                float var = lnS[5][row] * (1.f / HDIM) - mu * mu;
                float rs  = rsqrtf(var + EPSLN);
                float hb  = (hp[j][i] - mu) * rs * ghv[j] + bhv[j];
                h_bf[row * 512 + nb * 16 + mlow] = f2bf(hb);
                out[((size_t)t * BATCH + btile * 16 + row) * HDIM + nb * 16 + mlow] = hb;
            }
        }
        __syncthreads();
        if (tid < 96) lnS[tid >> 4][tid & 15] = 0.f;
        __syncthreads();
    }
}

// ---------------------------------------------------------------------------
extern "C" void kernel_launch(void* const* d_in, const int* in_sizes, int n_in,
                              void* d_out, int out_size, void* d_ws, size_t ws_size,
                              hipStream_t stream) {
    const float* x   = (const float*)d_in[0];
    const float* Wx  = (const float*)d_in[1];
    const float* bx  = (const float*)d_in[2];
    const float* Wh  = (const float*)d_in[3];
    const float* bh  = (const float*)d_in[4];
    const float* Wm  = (const float*)d_in[5];
    const float* bm  = (const float*)d_in[6];
    const float* Wpm = (const float*)d_in[7];
    const float* bpm = (const float*)d_in[8];
    const float* g_c = (const float*)d_in[9];
    const float* b_c = (const float*)d_in[10];
    const float* g_m = (const float*)d_in[11];
    const float* b_m = (const float*)d_in[12];
    const float* g_h = (const float*)d_in[13];
    const float* b_h = (const float*)d_in[14];

    // ws layout: [0, 7864320) packed bf16 weights ; [7864320, +167772160) xp tiles
    unsigned short* pack = (unsigned short*)d_ws;
    unsigned short* xp   = (unsigned short*)((char*)d_ws + 7864320);

    pack_weights_kernel<<<960, 256, 0, stream>>>(Wx, Wh, Wm, Wpm, pack);
    xproj_kernel<<<2048, 256, 0, stream>>>(x, pack, xp);
    recurrent_kernel<<<4, 512, 0, stream>>>(xp, pack, bx, bh, bm, bpm,
                                            g_c, b_c, g_m, b_m, g_h, b_h,
                                            (float*)d_out);
}
